// SSM8Block_68685116998060
// MI455X (gfx1250) — matrix operand order repaired and checked by reading, not compiled, not run on hardware
//
#include <hip/hip_runtime.h>
#include <hip/hip_bf16.h>

typedef __attribute__((ext_vector_type(16))) _Float16 v16h;
typedef __attribute__((ext_vector_type(8)))  _Float16 v8h;
typedef __attribute__((ext_vector_type(8)))  float    v8f;
typedef __attribute__((ext_vector_type(4)))  unsigned int u32x4;
typedef __attribute__((ext_vector_type(8)))  int          i32x8;
typedef __attribute__((ext_vector_type(4)))  int          i32x4;

#define D_MODEL 640
#define D_INNER 1280
#define D_STATE 34
#define DT_RANK 40
#define BATCH   8
#define SEQ     4096
#define MTOT    (BATCH*SEQ)          // 32768 rows
#define XZ_LD   (2*D_INNER)          // 2560
#define XBC_LD  128                  // 108 padded to 128
#define DT_KPAD 64                   // 40 padded to 64
#define KB      64                   // K-block staged per TDM transfer
#define TDM_LDS_BYTES (2 * 64 * KB * 2)   // double-buffered 64x64 f16 tile

__device__ __forceinline__ float softplusf(float x){
  return x > 20.f ? x : log1pf(__expf(x));
}

// ---------------- WMMA fragment loaders (f16 16x16x32, wave32) ----------------
// A: row-major [M,K]. Lane = hi*16 + m. VGPR v<4 -> K=hi*8+2v+p ; v>=4 -> K=16+hi*8+...
__device__ __forceinline__ v16h load_a_frag(const _Float16* A, int lda, int m0, int k0, int lane){
  const int hi = lane >> 4, m = lane & 15;
  const _Float16* p = A + (size_t)(m0 + m) * lda + (k0 + hi * 8);
  v8h r0 = *(const v8h*)(p);
  v8h r1 = *(const v8h*)(p + 16);
  v16h a;
#pragma unroll
  for (int i = 0; i < 8; ++i){ a[i] = r0[i]; a[8 + i] = r1[i]; }
  return a;
}
// B fragment from an LDS tile [64 rows x KB f16] (rows = output features n).
// Lane = hi*16 + n holds K = ks + hi*8 + 0..7 and ks + 16 + hi*8 + 0..7.
__device__ __forceinline__ v16h load_b_frag_lds(const _Float16* buf, int nloc, int ks, int lane){
  const int hi = lane >> 4, n = lane & 15;
  const _Float16* p = buf + (nloc + n) * KB + ks + hi * 8;
  v8h r0 = *(const v8h*)(p);
  v8h r1 = *(const v8h*)(p + 16);
  v16h b;
#pragma unroll
  for (int i = 0; i < 8; ++i){ b[i] = r0[i]; b[8 + i] = r1[i]; }
  return b;
}

// ---------------- Tensor Data Mover: stage a 64(rows) x KB(f16) weight tile to LDS ----------------
// D# per CDNA5 ISA. 2D tensor -> groups 2/3 zero.
__device__ __forceinline__ void tdm_issue_tile(const _Float16* W, int ldb, int n0, int k0,
                                               unsigned lds_off){
  unsigned long long ga = (unsigned long long)(uintptr_t)(W + (size_t)n0 * ldb + k0);
  u32x4 g0;
  g0[0] = 1u;                                        // count=1, is_restore=0, gather=0
  g0[1] = lds_off;                                   // LDS byte address (dynamic smem base = 0)
  g0[2] = (unsigned)(ga & 0xffffffffu);              // global_addr[31:0]
  g0[3] = (unsigned)((ga >> 32) & 0x01ffffffu)       // global_addr[56:32]
        | 0x80000000u;                               // type=2 ("image") in bits 127:126
  const int rows = 64, cols = KB;
  i32x8 g1;
  g1[0] = 0x00010000;                                // data_size=1 (2 bytes); no pad/iter/multicast
  g1[1] = (cols & 0xffff) << 16;                     // tensor_dim0[15:0]   (bits 63:48)
  g1[2] = ((cols >> 16) & 0xffff)                    // tensor_dim0[31:16]
        | ((rows & 0xffff) << 16);                   // tensor_dim1[15:0]
  g1[3] = ((rows >> 16) & 0xffff)                    // tensor_dim1[31:16]
        | ((cols & 0xffff) << 16);                   // tile_dim0 = KB
  g1[4] = rows & 0xffff;                             // tile_dim1 = 64, tile_dim2 = 0
  g1[5] = ldb;                                       // tensor_dim0_stride[31:0] (elements)
  g1[6] = 0;                                         // stride hi / dim1_stride lo
  g1[7] = 0;
  i32x4 z4 = {0, 0, 0, 0};
  i32x8 z8 = {0, 0, 0, 0, 0, 0, 0, 0};
  __builtin_amdgcn_tensor_load_to_lds(g0, g1, z4, z4, z8, 0);
}

// One wave accumulates a 16x64 strip; B tiles staged by TDM, double-buffered in LDS.
__device__ __forceinline__ void gemm_acc_16x64_tdm(const _Float16* A, const _Float16* W,
                                                   int K, int m0, int n0, v8f acc[4],
                                                   _Float16* smem){
  const int lane = threadIdx.x & 31;
  const int wave = threadIdx.x >> 5;
  const int nkb = K / KB;
  if (wave == 0) tdm_issue_tile(W, K, n0, 0, 0u);
  for (int kb = 0; kb < nkb; ++kb){
    if (wave == 0) __builtin_amdgcn_s_wait_tensorcnt(0);   // buf[kb] landed
    __syncthreads();                                       // release consumers
    if (wave == 0 && kb + 1 < nkb)                         // prefetch next tile into other buffer
      tdm_issue_tile(W, K, n0, (kb + 1) * KB, (unsigned)(((kb + 1) & 1) * (64 * KB * 2)));
    const _Float16* buf = smem + (size_t)(kb & 1) * (64 * KB);
    const int k0 = kb * KB;
#pragma unroll
    for (int ks = 0; ks < KB; ks += 32){
      v16h a = load_a_frag(A, K, m0, k0 + ks, lane);
      if (ks == 0 && kb + 1 < nkb)   // hint next A block (global_prefetch_b8)
        __builtin_prefetch((const void*)(A + (size_t)(m0 + (lane & 15)) * K + k0 + KB), 0, 1);
#pragma unroll
      for (int j = 0; j < 4; ++j){
        v16h b = load_b_frag_lds(buf, 16 * j, ks, lane);
        acc[j] = __builtin_amdgcn_wmma_f32_16x16x32_f16(false, a, false, b, (short)0, acc[j],
                                                        false, false);
      }
    }
    __syncthreads();                                       // all waves done with buf[kb]
  }
}

// ---------------- small prep kernels ----------------
__global__ __launch_bounds__(256) void k_rmsnorm(const float* __restrict__ x,
                                                 const float* __restrict__ w,
                                                 _Float16* __restrict__ out){
  const int m = blockIdx.x;
  const float* row = x + (size_t)m * D_MODEL;
  __shared__ float red[256];
  float s = 0.f;
  for (int k = threadIdx.x; k < D_MODEL; k += 256){ float v = row[k]; s += v * v; }
  red[threadIdx.x] = s; __syncthreads();
  for (int off = 128; off > 0; off >>= 1){
    if (threadIdx.x < off) red[threadIdx.x] += red[threadIdx.x + off];
    __syncthreads();
  }
  const float inv = rsqrtf(red[0] / (float)D_MODEL + 1e-6f);
  for (int k = threadIdx.x; k < D_MODEL; k += 256)
    out[(size_t)m * D_MODEL + k] = (_Float16)(row[k] * inv * w[k]);
}

__global__ __launch_bounds__(256) void k_cvt_f16(const float* __restrict__ in,
                                                 _Float16* __restrict__ out, int n){
  int i = blockIdx.x * 256 + threadIdx.x;
  if (i < n) out[i] = (_Float16)in[i];
}

// x_proj_w [108,1280] -> f16 [128,1280], rows 108..127 zero
__global__ __launch_bounds__(256) void k_cvt_xproj(const float* __restrict__ in,
                                                   _Float16* __restrict__ out){
  int i = blockIdx.x * 256 + threadIdx.x;           // over 128*1280
  if (i >= XBC_LD * D_INNER) return;
  int r = i / D_INNER, c = i % D_INNER;
  out[i] = (r < DT_RANK + 2 * D_STATE) ? (_Float16)in[(size_t)r * D_INNER + c] : (_Float16)0.f;
}

// dt_proj_w [1280,40] -> f16 [1280,64], cols 40..63 zero
__global__ __launch_bounds__(256) void k_cvt_dtproj(const float* __restrict__ in,
                                                    _Float16* __restrict__ out){
  int i = blockIdx.x * 256 + threadIdx.x;           // over 1280*64
  if (i >= D_INNER * DT_KPAD) return;
  int r = i / DT_KPAD, c = i % DT_KPAD;
  out[i] = (c < DT_RANK) ? (_Float16)in[(size_t)r * DT_RANK + c] : (_Float16)0.f;
}

// dt_raw (xbc cols 0..39) -> f16 padded [MTOT,64]
__global__ __launch_bounds__(256) void k_dtpad(const float* __restrict__ xbc,
                                               _Float16* __restrict__ out){
  size_t i = (size_t)blockIdx.x * 256 + threadIdx.x; // over MTOT*64
  if (i >= (size_t)MTOT * DT_KPAD) return;
  int c = (int)(i & (DT_KPAD - 1));
  size_t m = i >> 6;
  out[i] = (c < DT_RANK) ? (_Float16)xbc[m * XBC_LD + c] : (_Float16)0.f;
}

// depthwise causal conv K=4 over the x_inner half of xz, f16 out for next GEMM
__global__ __launch_bounds__(256) void k_conv(const float* __restrict__ xz,
                                              const float* __restrict__ cw,
                                              const float* __restrict__ cb,
                                              _Float16* __restrict__ out){
  size_t idx = (size_t)blockIdx.x * 256 + threadIdx.x;  // over MTOT*D_INNER
  if (idx >= (size_t)MTOT * D_INNER) return;
  int   c = (int)(idx % D_INNER);
  size_t m = idx / D_INNER;
  int   t = (int)(m % SEQ);
  size_t base = (m - t) * (size_t)XZ_LD + c;            // row b*SEQ, col c
  float acc = cb[c];
#pragma unroll
  for (int k = 0; k < 4; ++k){
    int tt = t - 3 + k;
    if (tt >= 0) acc += cw[c * 4 + k] * xz[base + (size_t)tt * XZ_LD];
  }
  out[idx] = (_Float16)acc;
}

// ---------------- WMMA GEMM kernels (block = 8 waves, 128x64 C-tile, TDM-staged B) ----------------
__global__ __launch_bounds__(256) void k_gemm_inproj(const _Float16* __restrict__ A,
                                                     const _Float16* __restrict__ B,
                                                     float* __restrict__ C){
  extern __shared__ _Float16 smem[];
  const int lane = threadIdx.x & 31, wave = threadIdx.x >> 5;
  const int m0 = blockIdx.y * 128 + wave * 16;
  const int n0 = blockIdx.x * 64;
  v8f acc[4] = {{}, {}, {}, {}};
  gemm_acc_16x64_tdm(A, B, D_MODEL, m0, n0, acc, smem);
  const int n = lane & 15, hi = lane >> 4;
#pragma unroll
  for (int j = 0; j < 4; ++j)
#pragma unroll
    for (int v = 0; v < 8; ++v)
      C[(size_t)(m0 + hi * 8 + v) * XZ_LD + n0 + 16 * j + n] = acc[j][v];
}

__global__ __launch_bounds__(256) void k_gemm_xproj(const _Float16* __restrict__ A,
                                                    const _Float16* __restrict__ B,
                                                    float* __restrict__ C){
  extern __shared__ _Float16 smem[];
  const int lane = threadIdx.x & 31, wave = threadIdx.x >> 5;
  const int m0 = blockIdx.y * 128 + wave * 16;
  const int n0 = blockIdx.x * 64;
  v8f acc[4] = {{}, {}, {}, {}};
  gemm_acc_16x64_tdm(A, B, D_INNER, m0, n0, acc, smem);
  const int n = lane & 15, hi = lane >> 4;
#pragma unroll
  for (int j = 0; j < 4; ++j)
#pragma unroll
    for (int v = 0; v < 8; ++v)
      C[(size_t)(m0 + hi * 8 + v) * XBC_LD + n0 + 16 * j + n] = acc[j][v];
}

// delta = softplus(softplus(dt_raw @ W^T + dt_proj_b) + dt_bias)
__global__ __launch_bounds__(256) void k_gemm_dt(const _Float16* __restrict__ A,
                                                 const _Float16* __restrict__ B,
                                                 const float* __restrict__ pb,
                                                 const float* __restrict__ db,
                                                 float* __restrict__ C){
  extern __shared__ _Float16 smem[];
  const int lane = threadIdx.x & 31, wave = threadIdx.x >> 5;
  const int m0 = blockIdx.y * 128 + wave * 16;
  const int n0 = blockIdx.x * 64;
  v8f acc[4] = {{}, {}, {}, {}};
  gemm_acc_16x64_tdm(A, B, DT_KPAD, m0, n0, acc, smem);
  const int n = lane & 15, hi = lane >> 4;
#pragma unroll
  for (int j = 0; j < 4; ++j){
    const int col = n0 + 16 * j + n;
    const float b0 = pb[col], b1 = db[col];
#pragma unroll
    for (int v = 0; v < 8; ++v){
      float d1 = softplusf(acc[j][v] + b0);
      C[(size_t)(m0 + hi * 8 + v) * D_INNER + col] = softplusf(d1 + b1);
    }
  }
}

// out = yg @ out_proj_w^T + residual(x)
__global__ __launch_bounds__(256) void k_gemm_out(const _Float16* __restrict__ A,
                                                  const _Float16* __restrict__ B,
                                                  const float* __restrict__ resid,
                                                  float* __restrict__ C){
  extern __shared__ _Float16 smem[];
  const int lane = threadIdx.x & 31, wave = threadIdx.x >> 5;
  const int m0 = blockIdx.y * 128 + wave * 16;
  const int n0 = blockIdx.x * 64;
  v8f acc[4] = {{}, {}, {}, {}};
  gemm_acc_16x64_tdm(A, B, D_INNER, m0, n0, acc, smem);
  const int n = lane & 15, hi = lane >> 4;
#pragma unroll
  for (int j = 0; j < 4; ++j)
#pragma unroll
    for (int v = 0; v < 8; ++v){
      size_t o = (size_t)(m0 + hi * 8 + v) * D_MODEL + n0 + 16 * j + n;
      C[o] = acc[j][v] + resid[o];
    }
}

// ---------------- selective scan: thread = (batch, channel), B/C staged in LDS ----------------
__global__ __launch_bounds__(256) void k_scan(const float* __restrict__ xz,
                                              const float* __restrict__ delta,
                                              const float* __restrict__ xbc,
                                              const float* __restrict__ A_log,
                                              const float* __restrict__ Dp,
                                              _Float16* __restrict__ yg){
  const int d = blockIdx.x * 256 + threadIdx.x;   // channel
  const int b = blockIdx.y;                       // batch
  __shared__ float sB[D_STATE], sC[D_STATE];
  float negA[D_STATE], h[D_STATE];
#pragma unroll
  for (int n = 0; n < D_STATE; ++n){
    negA[n] = -__expf(A_log[(size_t)d * D_STATE + n]);
    h[n] = 0.f;
  }
  const float Dpar = Dp[d];
  for (int t = 0; t < SEQ; ++t){
    const size_t m = (size_t)b * SEQ + t;
    __syncthreads();
    if (threadIdx.x < 2 * D_STATE){
      float v = xbc[m * XBC_LD + DT_RANK + threadIdx.x];
      if (threadIdx.x < D_STATE) sB[threadIdx.x] = v;
      else                       sC[threadIdx.x - D_STATE] = v;
    }
    __syncthreads();
    const float u  = xz[m * XZ_LD + d];
    const float dl = delta[m * D_INNER + d];
    const float z  = xz[m * XZ_LD + D_INNER + d];
    const float du = dl * u;
    float y = 0.f;
#pragma unroll
    for (int n = 0; n < D_STATE; ++n){
      h[n] = __expf(dl * negA[n]) * h[n] + du * sB[n];
      y += h[n] * sC[n];
    }
    y += u * Dpar;
    const float sz = z / (1.f + __expf(-z));
    y *= sz * sz;
    yg[m * D_INNER + d] = (_Float16)y;
  }
}

// ---------------- launch ----------------
extern "C" void kernel_launch(void* const* d_in, const int* in_sizes, int n_in,
                              void* d_out, int out_size, void* d_ws, size_t ws_size,
                              hipStream_t stream){
  const float* x        = (const float*)d_in[0];
  const float* norm_w   = (const float*)d_in[1];
  const float* in_proj  = (const float*)d_in[2];
  const float* conv_w   = (const float*)d_in[3];
  const float* conv_b   = (const float*)d_in[4];
  const float* x_proj   = (const float*)d_in[5];
  const float* dt_proj  = (const float*)d_in[6];
  const float* dt_pb    = (const float*)d_in[7];
  const float* A_log    = (const float*)d_in[8];
  const float* D_param  = (const float*)d_in[9];
  const float* dt_bias  = (const float*)d_in[10];
  const float* out_proj = (const float*)d_in[11];
  float* out = (float*)d_out;

  // workspace carve-up
  char* p = (char*)d_ws;
  auto take = [&](size_t bytes) -> char* {
    char* r = p; p += (bytes + 255) & ~(size_t)255; return r;
  };
  _Float16* xn_h    = (_Float16*)take((size_t)MTOT * D_MODEL * 2);
  _Float16* w_in_h  = (_Float16*)take((size_t)2 * D_INNER * D_MODEL * 2);
  _Float16* w_xp_h  = (_Float16*)take((size_t)XBC_LD * D_INNER * 2);
  _Float16* w_dt_h  = (_Float16*)take((size_t)D_INNER * DT_KPAD * 2);
  _Float16* w_out_h = (_Float16*)take((size_t)D_MODEL * D_INNER * 2);
  float*    xz      = (float*)   take((size_t)MTOT * XZ_LD * 4);
  _Float16* xconv_h = (_Float16*)take((size_t)MTOT * D_INNER * 2);
  float*    xbc     = (float*)   take((size_t)MTOT * XBC_LD * 4);
  _Float16* dtpad_h = (_Float16*)take((size_t)MTOT * DT_KPAD * 2);
  float*    delta   = (float*)   take((size_t)MTOT * D_INNER * 4);
  _Float16* yg_h    = (_Float16*)take((size_t)MTOT * D_INNER * 2);

  // 1. RMSNorm -> f16 activations
  k_rmsnorm<<<MTOT, 256, 0, stream>>>(x, norm_w, xn_h);

  // 2. weight conversions / paddings
  {
    int n1 = 2 * D_INNER * D_MODEL;
    k_cvt_f16<<<(n1 + 255) / 256, 256, 0, stream>>>(in_proj, w_in_h, n1);
    int n2 = D_MODEL * D_INNER;
    k_cvt_f16<<<(n2 + 255) / 256, 256, 0, stream>>>(out_proj, w_out_h, n2);
    int n3 = XBC_LD * D_INNER;
    k_cvt_xproj<<<(n3 + 255) / 256, 256, 0, stream>>>(x_proj, w_xp_h);
    int n4 = D_INNER * DT_KPAD;
    k_cvt_dtproj<<<(n4 + 255) / 256, 256, 0, stream>>>(dt_proj, w_dt_h);
  }

  // 3. in_proj GEMM: [32768,640] x [2560,640]^T -> xz [32768,2560]
  k_gemm_inproj<<<dim3(XZ_LD / 64, MTOT / 128), 256, TDM_LDS_BYTES, stream>>>(xn_h, w_in_h, xz);

  // 4. depthwise causal conv on x_inner half
  {
    size_t n = (size_t)MTOT * D_INNER;
    k_conv<<<(unsigned)((n + 255) / 256), 256, 0, stream>>>(xz, conv_w, conv_b, xconv_h);
  }

  // 5. x_proj GEMM: [32768,1280] x [128,1280]^T -> xbc [32768,128]
  k_gemm_xproj<<<dim3(XBC_LD / 64, MTOT / 128), 256, TDM_LDS_BYTES, stream>>>(xconv_h, w_xp_h, xbc);

  // 6. pack dt_raw -> f16 [32768,64]
  {
    size_t n = (size_t)MTOT * DT_KPAD;
    k_dtpad<<<(unsigned)((n + 255) / 256), 256, 0, stream>>>(xbc, dtpad_h);
  }

  // 7. dt_proj GEMM + double softplus epilogue -> delta [32768,1280]
  k_gemm_dt<<<dim3(D_INNER / 64, MTOT / 128), 256, TDM_LDS_BYTES, stream>>>(dtpad_h, w_dt_h,
                                                                            dt_pb, dt_bias, delta);

  // 8. selective scan + D*u + silu(z)^2 gating -> yg f16 [32768,1280]
  k_scan<<<dim3(D_INNER / 256, BATCH), 256, 0, stream>>>(xz, delta, xbc, A_log, D_param, yg_h);

  // 9. out_proj GEMM + residual -> d_out [32768,640]
  k_gemm_out<<<dim3(D_MODEL / 64, MTOT / 128), 256, TDM_LDS_BYTES, stream>>>(yg_h, w_out_h, x, out);

  (void)in_sizes; (void)n_in; (void)out_size; (void)ws_size;
}
